// NemotronFlashMambaDecoderLayer_7292854468795
// MI455X (gfx1250) — compile-verified, matrix operand order checked by reading, not run on hardware
//
#include <hip/hip_runtime.h>
#include <hip/hip_bf16.h>

// ---------------------------------------------------------------------------
// Dims
// ---------------------------------------------------------------------------
#define Hd     1024
#define Ed     2048
#define NHd    32
#define Pd     64
#define Nd_    128
#define Kc     4
#define Qd     128
#define FFd    4096
#define CONVD  2304
#define DIN    4384
#define Bb     2
#define Ld     2048
#define BL     4096      /* Bb*Ld */
#define CH     16        /* Ld/Qd chunks per batch */

typedef __bf16 bf16;
typedef __attribute__((ext_vector_type(16))) __bf16 v16bf;
typedef __attribute__((ext_vector_type(8)))  __bf16 v8bf;
typedef __attribute__((ext_vector_type(8)))  float  v8f;

__device__ inline v8f v8f_zero() {
  v8f v;
#pragma unroll
  for (int i = 0; i < 8; ++i) v[i] = 0.f;
  return v;
}

// Async 16B global -> LDS copy (CDNA5 GLOBAL_LOAD_ASYNC_TO_LDS_B128,
// tracked by ASYNCcnt). LDS byte address = low 32 bits of the generic
// pointer (ISA 10.2: LDS aperture truncates addr[31:0]).
__device__ inline void async_cp16(void* lds_dst, const void* gsrc) {
  unsigned ldsa = (unsigned)(size_t)lds_dst;
  asm volatile("global_load_async_to_lds_b128 %0, %1, off"
               :: "v"(ldsa), "v"(gsrc) : "memory");
}
__device__ inline void wait_async0() {
  asm volatile("s_wait_asynccnt 0x0" ::: "memory");
}

// Load a 16x32 bf16 fragment (A or B operand) from row-major storage with
// leading dimension ld (elements). Matches the CDNA5 16-bit A/B VGPR layout:
// lane<16 holds row=lane, K {0..7, 16..23}; lane>=16 holds row=lane-16,
// K {8..15, 24..31}. Rows must be 16B aligned.
__device__ inline v16bf ldsFrag(const bf16* base, int ld) {
  int lane = threadIdx.x & 31;
  const bf16* r = base + (lane & 15) * ld + ((lane >> 4) << 3);
  v8bf lo = *(const v8bf*)(r);
  v8bf hi = *(const v8bf*)(r + 16);
  v16bf f;
#pragma unroll
  for (int i = 0; i < 8; ++i) { f[i] = lo[i]; f[i + 8] = hi[i]; }
  return f;
}

__device__ inline v8f wmma_bf16(v16bf a, v16bf b, v8f c) {
  return __builtin_amdgcn_wmma_f32_16x16x32_bf16(false, a, false, b, (short)0, c,
                                                 false, false);
}

__device__ inline float sigmoidf_(float x) { return 1.f / (1.f + __expf(-x)); }

// ---------------------------------------------------------------------------
// Generic bf16 GEMM:  C[m,n] = sum_k A[m,k] * W[n,k]  (+ optional addend)
// Block tile 128x128, 256 threads = 8 waves, each wave a 16-row strip x 128.
// Double-buffered async global->LDS tile staging overlapped with WMMA.
// M must be a multiple of 128 (true for every call site); N is guarded.
// ---------------------------------------------------------------------------
__global__ __launch_bounds__(256)
void gemm_bf16_nt(const bf16* __restrict__ A, const bf16* __restrict__ W,
                  float* __restrict__ C, const float* __restrict__ addend,
                  int M, int Nn, int Kd,
                  long batchA, long batchW, long batchC)
{
  __shared__ bf16 As[2][128 * 40];
  __shared__ bf16 Ws[2][128 * 40];

  long z = blockIdx.z;
  A += z * batchA;
  W += z * batchW;
  C += z * batchC;
  if (addend) addend += z * batchC;

  const int n0 = blockIdx.x * 128, m0 = blockIdx.y * 128;
  const int t = threadIdx.x, w = t >> 5, lane = t & 31;
  const int row = t >> 1, col = (t & 1) << 4;
  const int gnRow = n0 + row;
  const bool wOk = gnRow < Nn;

  v8f acc[8];
#pragma unroll
  for (int i = 0; i < 8; ++i) acc[i] = v8f_zero();

  auto loadTile = [&](int k0, int buf) {
    // A tile: always in range (M multiple of 128).
    const bf16* sa = A + (size_t)(m0 + row) * Kd + k0 + col;
    bf16* da = As[buf] + row * 40 + col;
    async_cp16(da, sa);
    async_cp16(da + 8, sa + 8);
    // W tile: guard rows, zero-fill OOB so masked columns stay finite.
    bf16* dw = Ws[buf] + row * 40 + col;
    if (wOk) {
      const bf16* sw = W + (size_t)gnRow * Kd + k0 + col;
      async_cp16(dw, sw);
      async_cp16(dw + 8, sw + 8);
    } else {
      uint4 zz = make_uint4(0, 0, 0, 0);
      *(uint4*)dw = zz;
      *(uint4*)(dw + 8) = zz;
    }
  };

  const int nk = Kd >> 5;
  loadTile(0, 0);
  wait_async0();
  __syncthreads();

  for (int ki = 0; ki < nk; ++ki) {
    const int buf = ki & 1;
    if (ki + 1 < nk) loadTile((ki + 1) << 5, buf ^ 1);
    if (ki + 2 < nk) {
      __builtin_prefetch(A + (size_t)(m0 + row) * Kd + ((ki + 2) << 5) + col, 0, 1);
      if (wOk)
        __builtin_prefetch(W + (size_t)gnRow * Kd + ((ki + 2) << 5) + col, 0, 1);
    }
    v16bf af = ldsFrag(As[buf] + (w << 4) * 40, 40);
#pragma unroll
    for (int nt = 0; nt < 8; ++nt) {
      v16bf bfr = ldsFrag(Ws[buf] + (nt << 4) * 40, 40);
      acc[nt] = wmma_bf16(af, bfr, acc[nt]);
    }
    wait_async0();
    __syncthreads();
  }

  const int nIn = lane & 15, mh = (lane >> 4) << 3;
  const int gmBase = m0 + (w << 4) + mh;
#pragma unroll
  for (int nt = 0; nt < 8; ++nt) {
    int gn = n0 + (nt << 4) + nIn;
    if (gn < Nn) {
#pragma unroll
      for (int r = 0; r < 8; ++r) {
        int gm = gmBase + r;
        float v = acc[nt][r];
        if (addend) v += addend[(size_t)gm * Nn + gn];
        C[(size_t)gm * Nn + gn] = v;
      }
    }
  }
}

// ---------------------------------------------------------------------------
// Elementwise / norm kernels
// ---------------------------------------------------------------------------
__global__ __launch_bounds__(256)
void cvt_f32_bf16(const float* __restrict__ in, bf16* __restrict__ out, long n)
{
  for (long i = (long)blockIdx.x * 256 + threadIdx.x; i < n;
       i += (long)gridDim.x * 256)
    out[i] = (bf16)in[i];
}

__global__ __launch_bounds__(256)
void rmsnorm_bf16(const float* __restrict__ X, const float* __restrict__ Wt,
                  bf16* __restrict__ Out, int D, float eps)
{
  __shared__ float red[256];
  const int row = blockIdx.x;
  const float* x = X + (size_t)row * D;
  float ss = 0.f;
  for (int i = threadIdx.x; i < D; i += 256) { float v = x[i]; ss += v * v; }
  red[threadIdx.x] = ss;
  __syncthreads();
  for (int o = 128; o > 0; o >>= 1) {
    if (threadIdx.x < o) red[threadIdx.x] += red[threadIdx.x + o];
    __syncthreads();
  }
  float rstd = rsqrtf(red[0] / (float)D + eps);
  for (int i = threadIdx.x; i < D; i += 256)
    Out[(size_t)row * D + i] = (bf16)(x[i] * rstd * Wt[i]);
}

// Causal depthwise conv (K=4) + SiLU over the xBC channels of zxbcdt.
__global__ __launch_bounds__(256)
void conv_silu(const float* __restrict__ zx, const float* __restrict__ wconv,
               const float* __restrict__ bconv, float* __restrict__ xc)
{
  size_t i = (size_t)blockIdx.x * 256 + threadIdx.x;
  if (i >= (size_t)BL * CONVD) return;
  int ch = (int)(i % CONVD);
  size_t bl = i / CONVD;
  int l = (int)(bl % Ld);
  int bb = (int)(bl / Ld);
  float acc = bconv[ch];
#pragma unroll
  for (int k = 0; k < Kc; ++k) {
    int lk = l - (Kc - 1) + k;
    if (lk >= 0)
      acc += zx[((size_t)(bb * Ld + lk)) * DIN + Ed + ch] * wconv[ch * Kc + k];
  }
  xc[i] = acc * sigmoidf_(acc);
}

__global__ __launch_bounds__(256)
void dt_softplus(const float* __restrict__ zx, const float* __restrict__ dt_bias,
                 float* __restrict__ dt)
{
  size_t i = (size_t)blockIdx.x * 256 + threadIdx.x;
  if (i >= (size_t)BL * NHd) return;
  int h = (int)(i & (NHd - 1));
  size_t bl = i >> 5;
  float x = zx[bl * DIN + Ed + CONVD + h] + dt_bias[h];
  dt[i] = (x > 20.f) ? x : log1pf(__expf(x));
}

// xdbf[b,l,h,p] = bf16(x*dt); Bbf/Cbf = bf16 copies of B/C channels.
__global__ __launch_bounds__(256)
void ssd_prep(const float* __restrict__ xc, const float* __restrict__ dt,
              bf16* __restrict__ xdbf, bf16* __restrict__ Bbf,
              bf16* __restrict__ Cbf)
{
  size_t i = (size_t)blockIdx.x * 256 + threadIdx.x;
  if (i < (size_t)BL * Ed) {
    int e = (int)(i & (Ed - 1));
    size_t bl = i >> 11;
    xdbf[i] = (bf16)(xc[bl * CONVD + e] * dt[bl * NHd + (e >> 6)]);
  }
  if (i < (size_t)BL * Nd_) {
    int n = (int)(i & (Nd_ - 1));
    size_t bl = i >> 7;
    Bbf[i] = (bf16)xc[bl * CONVD + Ed + n];
    Cbf[i] = (bf16)xc[bl * CONVD + Ed + Nd_ + n];
  }
}

// Per (b,h,c): inclusive scan of dA over q -> Acs.
__global__ __launch_bounds__(128)
void ssd_scan(const float* __restrict__ dt, const float* __restrict__ A_log,
              float* __restrict__ Acs)
{
  __shared__ float s[128];
  int c = blockIdx.x, h = blockIdx.y, b = blockIdx.z, q = threadIdx.x;
  float A = -__expf(A_log[h]);
  s[q] = dt[((size_t)(b * Ld + c * Qd + q)) * NHd + h] * A;
  __syncthreads();
  for (int off = 1; off < 128; off <<= 1) {
    float add = (q >= off) ? s[q - off] : 0.f;
    __syncthreads();
    s[q] += add;
    __syncthreads();
  }
  Acs[(((size_t)(b * NHd + h) * CH + c) << 7) + q] = s[q];
}

// ---------------------------------------------------------------------------
// SSD WMMA kernels
// ---------------------------------------------------------------------------
// Y_diag: per (b,c,h): A2[q,s] = (s<=q) ? G[q,s]*exp(acs[q]-acs[s]) : 0 (bf16)
//         Y[q,p] = sum_s A2[q,s] * xd[s,p]
__global__ __launch_bounds__(256)
void ssd_ydiag(const float* __restrict__ G, const float* __restrict__ Acs,
               const bf16* __restrict__ xdbf, float* __restrict__ Y)
{
  __shared__ float acs_s[128];
  __shared__ bf16 A2[128 * 136];
  __shared__ bf16 xdT[64 * 136];
  const int h = blockIdx.x, c = blockIdx.y, b = blockIdx.z;
  const int t = threadIdx.x, w = t >> 5, lane = t & 31;

  const float* acsp = Acs + (((size_t)(b * NHd + h) * CH + c) << 7);
  if (t < 128) acs_s[t] = acsp[t];
  __syncthreads();

  const float* Gp = G + ((size_t)(b * CH + c) << 14);
  for (int i = t; i < 128 * 128; i += 256) {
    int q = i >> 7, s = i & 127;
    float v = (s <= q) ? Gp[i] * __expf(acs_s[q] - acs_s[s]) : 0.f;
    A2[q * 136 + s] = (bf16)v;
  }
  const bf16* xp = xdbf + ((size_t)(b * Ld + c * Qd) * NHd + h) * (size_t)Pd;
  for (int i = t; i < 128 * 64; i += 256) {
    int s = i >> 6, p = i & 63;
    xdT[p * 136 + s] = xp[(size_t)s * (NHd * Pd) + p];
  }
  __syncthreads();

  v8f acc[4];
#pragma unroll
  for (int i = 0; i < 4; ++i) acc[i] = v8f_zero();
  for (int s0 = 0; s0 < 128; s0 += 32) {
    v16bf af = ldsFrag(A2 + (w << 4) * 136 + s0, 136);
#pragma unroll
    for (int pt = 0; pt < 4; ++pt) {
      v16bf bfr = ldsFrag(xdT + (pt << 4) * 136 + s0, 136);
      acc[pt] = wmma_bf16(af, bfr, acc[pt]);
    }
  }
  const int nIn = lane & 15, mh = (lane >> 4) << 3;
#pragma unroll
  for (int pt = 0; pt < 4; ++pt) {
#pragma unroll
    for (int r = 0; r < 8; ++r) {
      int q = (w << 4) + mh + r;
      int p = (pt << 4) + nIn;
      Y[((size_t)(b * Ld + c * Qd + q) * NHd + h) * Pd + p] = acc[pt][r];
    }
  }
}

// states: per (b,c,h): S[p,n] = sum_q xd[q,p]*exp(acs[127]-acs[q]) * B[q,n]
__global__ __launch_bounds__(256)
void ssd_states(const bf16* __restrict__ xdbf, const bf16* __restrict__ Bbf,
                const float* __restrict__ Acs, float* __restrict__ S)
{
  __shared__ float acs_s[128];
  __shared__ bf16 xdT[64 * 136];
  __shared__ bf16 BT[128 * 136];
  const int h = blockIdx.x, c = blockIdx.y, b = blockIdx.z;
  const int t = threadIdx.x, w = t >> 5, lane = t & 31;

  const float* acsp = Acs + (((size_t)(b * NHd + h) * CH + c) << 7);
  if (t < 128) acs_s[t] = acsp[t];
  __syncthreads();
  const float alast = acs_s[127];

  const bf16* xp = xdbf + ((size_t)(b * Ld + c * Qd) * NHd + h) * (size_t)Pd;
  for (int i = t; i < 128 * 64; i += 256) {
    int s = i >> 6, p = i & 63;
    float xv = (float)xp[(size_t)s * (NHd * Pd) + p];
    xdT[p * 136 + s] = (bf16)(xv * __expf(alast - acs_s[s]));
  }
  const bf16* Bp = Bbf + (size_t)(b * Ld + c * Qd) * Nd_;
  for (int i = t; i < 128 * 128; i += 256) {
    int q = i >> 7, n = i & 127;
    BT[n * 136 + q] = Bp[i];
  }
  __syncthreads();

  const int mt = w & 3, nb = (w >> 2) << 2;
  v8f acc[4];
#pragma unroll
  for (int i = 0; i < 4; ++i) acc[i] = v8f_zero();
  for (int q0 = 0; q0 < 128; q0 += 32) {
    v16bf af = ldsFrag(xdT + (mt << 4) * 136 + q0, 136);
#pragma unroll
    for (int j = 0; j < 4; ++j) {
      v16bf bfr = ldsFrag(BT + ((nb + j) << 4) * 136 + q0, 136);
      acc[j] = wmma_bf16(af, bfr, acc[j]);
    }
  }
  const int nIn = lane & 15, mh = (lane >> 4) << 3;
  size_t base = ((size_t)((b * CH + c) * NHd + h)) << 13;  // *Pd*Nd_
#pragma unroll
  for (int j = 0; j < 4; ++j) {
#pragma unroll
    for (int r = 0; r < 8; ++r) {
      int p = (mt << 4) + mh + r;
      int n = ((nb + j) << 4) + nIn;
      S[base + (size_t)p * Nd_ + n] = acc[j][r];
    }
  }
}

// Serial inter-chunk recurrence: NS[z] = S_z, S_z = exp(alast[z-1])*S_{z-1}+states[z-1]
__global__ __launch_bounds__(256)
void ssd_chunkscan(const float* __restrict__ S, const float* __restrict__ Acs,
                   float* __restrict__ NS)
{
  size_t i = (size_t)blockIdx.x * 256 + threadIdx.x;
  if (i >= (size_t)Bb * NHd * Pd * Nd_) return;
  int n = (int)(i & 127);
  int p = (int)((i >> 7) & 63);
  int h = (int)((i >> 13) & 31);
  int b = (int)(i >> 18);
  float s = 0.f;
  for (int c = 0; c < CH; ++c) {
    size_t idx = (((size_t)((b * CH + c) * NHd + h)) << 13) + ((size_t)p << 7) + n;
    NS[idx] = s;
    float alast = Acs[(((size_t)(b * NHd + h) * CH + c) << 7) + 127];
    s = s * __expf(alast) + S[idx];
  }
}

// Y_off: per (b,c,h): Y[q,p] += exp(acs[q]) * sum_n C[q,n]*NS[p,n]
__global__ __launch_bounds__(256)
void ssd_yoff(const bf16* __restrict__ Cbf, const float* __restrict__ NS,
              const float* __restrict__ Acs, float* __restrict__ Y)
{
  __shared__ float acs_s[128];
  __shared__ bf16 Cs[128 * 136];
  __shared__ bf16 NSs[64 * 136];
  const int h = blockIdx.x, c = blockIdx.y, b = blockIdx.z;
  const int t = threadIdx.x, w = t >> 5, lane = t & 31;

  const float* acsp = Acs + (((size_t)(b * NHd + h) * CH + c) << 7);
  if (t < 128) acs_s[t] = acsp[t];

  // C tile: straight row-major copy -> async global->LDS (16B chunks).
  const bf16* Cp = Cbf + (size_t)(b * Ld + c * Qd) * Nd_;
  for (int i = t * 8; i < 128 * 128; i += 256 * 8) {
    int q = i >> 7, n = i & 127;
    async_cp16(Cs + q * 136 + n, Cp + i);
  }
  size_t nsb = ((size_t)((b * CH + c) * NHd + h)) << 13;
  for (int i = t; i < 64 * 128; i += 256) {
    int p = i >> 7, n = i & 127;
    NSs[p * 136 + n] = (bf16)NS[nsb + i];
  }
  wait_async0();
  __syncthreads();

  v8f acc[4];
#pragma unroll
  for (int i = 0; i < 4; ++i) acc[i] = v8f_zero();
  for (int n0 = 0; n0 < 128; n0 += 32) {
    v16bf af = ldsFrag(Cs + (w << 4) * 136 + n0, 136);
#pragma unroll
    for (int pt = 0; pt < 4; ++pt) {
      v16bf bfr = ldsFrag(NSs + (pt << 4) * 136 + n0, 136);
      acc[pt] = wmma_bf16(af, bfr, acc[pt]);
    }
  }
  const int nIn = lane & 15, mh = (lane >> 4) << 3;
#pragma unroll
  for (int pt = 0; pt < 4; ++pt) {
#pragma unroll
    for (int r = 0; r < 8; ++r) {
      int q = (w << 4) + mh + r;
      int p = (pt << 4) + nIn;
      size_t idx = ((size_t)(b * Ld + c * Qd + q) * NHd + h) * Pd + p;
      Y[idx] += __expf(acs_s[q]) * acc[pt][r];
    }
  }
}

// Gated RMSNorm: g = (Y + D[h]*x) * silu(z); out = rmsnorm(g)*w_mnorm (bf16)
__global__ __launch_bounds__(256)
void gated_norm(const float* __restrict__ Y, const float* __restrict__ xc,
                const float* __restrict__ zx, const float* __restrict__ Dp,
                const float* __restrict__ wm, bf16* __restrict__ Out)
{
  __shared__ float buf[Ed];
  __shared__ float red[256];
  const int row = blockIdx.x;
  const float* y = Y + (size_t)row * Ed;
  const float* x = xc + (size_t)row * CONVD;
  const float* z = zx + (size_t)row * DIN;
  float ss = 0.f;
  for (int e = threadIdx.x; e < Ed; e += 256) {
    float yv = y[e] + Dp[e >> 6] * x[e];
    float zv = z[e];
    float g = yv * (zv * sigmoidf_(zv));
    buf[e] = g;
    ss += g * g;
  }
  red[threadIdx.x] = ss;
  __syncthreads();
  for (int o = 128; o > 0; o >>= 1) {
    if (threadIdx.x < o) red[threadIdx.x] += red[threadIdx.x + o];
    __syncthreads();
  }
  float rstd = rsqrtf(red[0] / (float)Ed + 1e-5f);
  for (int e = threadIdx.x; e < Ed; e += 256)
    Out[(size_t)row * Ed + e] = (bf16)(buf[e] * rstd * wm[e]);
}

__global__ __launch_bounds__(256)
void silu_mul_bf16(const float* __restrict__ g, const float* __restrict__ u,
                   bf16* __restrict__ out, long n)
{
  long i = (long)blockIdx.x * 256 + threadIdx.x;
  if (i >= n) return;
  float x = g[i];
  out[i] = (bf16)(x * sigmoidf_(x) * u[i]);
}

// ---------------------------------------------------------------------------
// Host launch
// ---------------------------------------------------------------------------
extern "C" void kernel_launch(void* const* d_in, const int* in_sizes, int n_in,
                              void* d_out, int out_size, void* d_ws, size_t ws_size,
                              hipStream_t stream)
{
  const float* hidden  = (const float*)d_in[0];
  const float* w_ln1   = (const float*)d_in[1];
  const float* w_in    = (const float*)d_in[2];
  const float* w_conv  = (const float*)d_in[3];
  const float* b_conv  = (const float*)d_in[4];
  const float* dt_bias = (const float*)d_in[5];
  const float* A_log   = (const float*)d_in[6];
  const float* Dp      = (const float*)d_in[7];
  const float* w_mnorm = (const float*)d_in[8];
  const float* w_out   = (const float*)d_in[9];
  const float* w_ln2   = (const float*)d_in[10];
  const float* w_gate  = (const float*)d_in[11];
  const float* w_up    = (const float*)d_in[12];
  const float* w_down  = (const float*)d_in[13];

  char* ws = (char*)d_ws;
  size_t off = 0;
  auto alloc = [&](size_t bytes) -> char* {
    char* p = ws + off;
    off += (bytes + 255) & ~(size_t)255;
    return p;
  };

  bf16*  hbf    = (bf16*)alloc((size_t)BL * Hd * 2);
  bf16*  winb   = (bf16*)alloc((size_t)DIN * Hd * 2);
  bf16*  woutb  = (bf16*)alloc((size_t)Hd * Ed * 2);
  bf16*  wgateb = (bf16*)alloc((size_t)FFd * Hd * 2);
  bf16*  wupb   = (bf16*)alloc((size_t)FFd * Hd * 2);
  bf16*  wdownb = (bf16*)alloc((size_t)Hd * FFd * 2);
  float* zxbcdt = (float*)alloc((size_t)BL * DIN * 4);      // aliased by gbuf
  float* xconv  = (float*)alloc((size_t)BL * CONVD * 4);    // aliased by ubuf
  float* dtbuf  = (float*)alloc((size_t)BL * NHd * 4);
  float* acs    = (float*)alloc((size_t)Bb * NHd * CH * Qd * 4);
  bf16*  xdbf   = (bf16*)alloc((size_t)BL * Ed * 2);
  bf16*  Bbf    = (bf16*)alloc((size_t)BL * Nd_ * 2);
  bf16*  Cbf    = (bf16*)alloc((size_t)BL * Nd_ * 2);
  float* Gbuf   = (float*)alloc((size_t)Bb * CH * Qd * Qd * 4);
  float* states = (float*)alloc((size_t)Bb * CH * NHd * Pd * Nd_ * 4);
  float* nstate = (float*)alloc((size_t)Bb * CH * NHd * Pd * Nd_ * 4); // aliased by abf
  float* Ybuf   = (float*)alloc((size_t)BL * Ed * 4);
  bf16*  ybf    = (bf16*)alloc((size_t)BL * Ed * 2);
  float* hid1   = (float*)alloc((size_t)BL * Hd * 4);
  bf16*  h2bf   = (bf16*)alloc((size_t)BL * Hd * 2);

  // Phase-2 aliases over dead phase-1 regions.
  float* gbuf = zxbcdt;           // 67.1MB <= 71.8MB
  float* ubuf = xconv;            // 67.1MB spans xconv..states (all dead by then)
  bf16*  abf  = (bf16*)nstate;    // 33.5MB == nstates size

  // 1) Weight conversions to bf16.
  cvt_f32_bf16<<<4096, 256, 0, stream>>>(w_in,   winb,   (long)DIN * Hd);
  cvt_f32_bf16<<<4096, 256, 0, stream>>>(w_out,  woutb,  (long)Hd * Ed);
  cvt_f32_bf16<<<4096, 256, 0, stream>>>(w_gate, wgateb, (long)FFd * Hd);
  cvt_f32_bf16<<<4096, 256, 0, stream>>>(w_up,   wupb,   (long)FFd * Hd);
  cvt_f32_bf16<<<4096, 256, 0, stream>>>(w_down, wdownb, (long)Hd * FFd);

  // 2) RMSNorm 1 -> bf16.
  rmsnorm_bf16<<<BL, 256, 0, stream>>>(hidden, w_ln1, hbf, Hd, 1e-6f);

  // 3) in_proj: [BL,Hd] x [DIN,Hd]^T -> zxbcdt.
  gemm_bf16_nt<<<dim3((DIN + 127) / 128, BL / 128, 1), 256, 0, stream>>>(
      hbf, winb, zxbcdt, nullptr, BL, DIN, Hd, 0, 0, 0);

  // 4) conv + silu; 5) dt softplus; 6) prep xd/B/C; 7) cumsum scan.
  conv_silu<<<(BL * CONVD) / 256, 256, 0, stream>>>(zxbcdt, w_conv, b_conv, xconv);
  dt_softplus<<<(BL * NHd) / 256, 256, 0, stream>>>(zxbcdt, dt_bias, dtbuf);
  ssd_prep<<<((size_t)BL * Ed) / 256, 256, 0, stream>>>(xconv, dtbuf, xdbf, Bbf, Cbf);
  ssd_scan<<<dim3(CH, NHd, Bb), 128, 0, stream>>>(dtbuf, A_log, acs);

  // 8) G = C . B^T per (b,c) — batched WMMA GEMM, 128x128x128.
  gemm_bf16_nt<<<dim3(1, 1, Bb * CH), 256, 0, stream>>>(
      Cbf, Bbf, Gbuf, nullptr, Qd, Qd, Nd_,
      (long)Qd * Nd_, (long)Qd * Nd_, (long)Qd * Qd);

  // 9) Y_diag; 10) chunk states; 11) inter-chunk scan; 12) Y_off.
  ssd_ydiag<<<dim3(NHd, CH, Bb), 256, 0, stream>>>(Gbuf, acs, xdbf, Ybuf);
  ssd_states<<<dim3(NHd, CH, Bb), 256, 0, stream>>>(xdbf, Bbf, acs, states);
  ssd_chunkscan<<<((size_t)Bb * NHd * Pd * Nd_) / 256, 256, 0, stream>>>(states, acs, nstate);
  ssd_yoff<<<dim3(NHd, CH, Bb), 256, 0, stream>>>(Cbf, nstate, acs, Ybuf);

  // 13) skip + gating + gated RMSNorm -> bf16.
  gated_norm<<<BL, 256, 0, stream>>>(Ybuf, xconv, zxbcdt, Dp, w_mnorm, ybf);

  // 14) out_proj + residual -> hid1.
  gemm_bf16_nt<<<dim3(Hd / 128, BL / 128, 1), 256, 0, stream>>>(
      ybf, woutb, hid1, hidden, BL, Hd, Ed, 0, 0, 0);

  // 15) RMSNorm 2 -> bf16.
  rmsnorm_bf16<<<BL, 256, 0, stream>>>(hid1, w_ln2, h2bf, Hd, 1e-6f);

  // 16/17) gate & up GEMMs.
  gemm_bf16_nt<<<dim3(FFd / 128, BL / 128, 1), 256, 0, stream>>>(
      h2bf, wgateb, gbuf, nullptr, BL, FFd, Hd, 0, 0, 0);
  gemm_bf16_nt<<<dim3(FFd / 128, BL / 128, 1), 256, 0, stream>>>(
      h2bf, wupb, ubuf, nullptr, BL, FFd, Hd, 0, 0, 0);

  // 18) silu(gate)*up -> bf16.
  silu_mul_bf16<<<((size_t)BL * FFd) / 256, 256, 0, stream>>>(
      gbuf, ubuf, abf, (long)BL * FFd);

  // 19) down GEMM + residual -> d_out (f32).
  gemm_bf16_nt<<<dim3(Hd / 128, BL / 128, 1), 256, 0, stream>>>(
      abf, wdownb, (float*)d_out, hid1, BL, Hd, FFd, 0, 0, 0);

  (void)in_sizes; (void)n_in; (void)out_size; (void)ws_size;
}
